// CausalSelfAttention_22608707846952
// MI455X (gfx1250) — matrix operand order repaired and checked by reading, not compiled, not run on hardware
//
#include <hip/hip_runtime.h>
#include <hip/hip_bf16.h>

#define Bb 4
#define Tt 2048
#define Cc 1024
#define Hh 16
#define DHd 64
#define BT (Bb * Tt)      /* 8192 */
#define FF (3 * Cc)       /* 3072 */

typedef __attribute__((ext_vector_type(16))) __bf16 v16bf;
typedef __attribute__((ext_vector_type(8)))  __bf16 v8bf;
typedef __attribute__((ext_vector_type(8)))  float  v8f;
typedef unsigned int v4u __attribute__((ext_vector_type(4)));
typedef int v8i_ __attribute__((ext_vector_type(8)));
typedef int v4i_ __attribute__((ext_vector_type(4)));

static __device__ __forceinline__ v8f zero8() {
  v8f z = {0.f, 0.f, 0.f, 0.f, 0.f, 0.f, 0.f, 0.f};
  return z;
}

static __device__ __forceinline__ v8f wmma_bf16(v16bf a, v16bf b, v8f c) {
  // D = A(16x32 bf16) * B(32x16 bf16) + C(16x16 f32)
  return __builtin_amdgcn_wmma_f32_16x16x32_bf16(
      /*neg_a=*/false, a, /*neg_b=*/false, b,
      /*c_mod=*/(short)0, c, /*reuse_a=*/false, /*reuse_b=*/false);
}

// A-matrix 16x32 bf16 fragment: lane m=l&15, hk=l>>4 holds
// K = hk*16+0..15 contiguous, row M = m.
static __device__ __forceinline__ v16bf load_a16x32(const __bf16* base, int ld, int lane) {
  const int m  = lane & 15;
  const int hk = (lane >> 4) & 1;
  const __bf16* p = base + (size_t)m * ld + hk * 16;
  v8bf lo = *(const v8bf*)p;
  v8bf hi = *(const v8bf*)(p + 8);
  return __builtin_shufflevector(lo, hi, 0,1,2,3,4,5,6,7,8,9,10,11,12,13,14,15);
}

// B-matrix 32x16 bf16 fragment: lane column n=l&15 holds K = hk*16+0..15
// contiguous. `base` points at (K=0, N=n0) of a source stored [N][K], ld = K-stride.
static __device__ __forceinline__ v16bf load_b32x16(const __bf16* base, int ld, int lane) {
  const int n  = lane & 15;
  const int hk = (lane >> 4) & 1;
  const __bf16* p = base + (size_t)n * ld + hk * 16;
  v8bf lo = *(const v8bf*)p;
  v8bf hi = *(const v8bf*)(p + 8);
  return __builtin_shufflevector(lo, hi, 0,1,2,3,4,5,6,7,8,9,10,11,12,13,14,15);
}

static __device__ __forceinline__ float rmax16(float v) {
  v = fmaxf(v, __shfl_xor(v, 1, 32));
  v = fmaxf(v, __shfl_xor(v, 2, 32));
  v = fmaxf(v, __shfl_xor(v, 4, 32));
  v = fmaxf(v, __shfl_xor(v, 8, 32));
  return v;
}
static __device__ __forceinline__ float rsum16(float v) {
  v += __shfl_xor(v, 1, 32);
  v += __shfl_xor(v, 2, 32);
  v += __shfl_xor(v, 4, 32);
  v += __shfl_xor(v, 8, 32);
  return v;
}

// ---------------------------------------------------------------------------
// Tensor Data Mover: DMA a 2D tile (tile_w elems of 2B per row, tile_h rows,
// row stride `stride_elems`) from global into LDS at byte offset lds_off.
// D# packing per CDNA5 ISA ch.8: group0 = {count, lds_addr, global_addr, type=2},
// group1 = {data_size=2B, tensor_dim0/1, tile_dim0/1, dim0_stride}.
// ---------------------------------------------------------------------------
static __device__ __forceinline__ void tdm_load_2d(unsigned lds_off, const void* gsrc,
                                                   unsigned tile_w, unsigned tile_h,
                                                   unsigned long long stride_elems) {
  unsigned long long ga = (unsigned long long)gsrc;
  unsigned ga_lo = __builtin_amdgcn_readfirstlane((unsigned)ga);
  unsigned ga_hi = __builtin_amdgcn_readfirstlane((unsigned)(ga >> 32));
  lds_off = __builtin_amdgcn_readfirstlane(lds_off);
  v4u g0;
  g0.x = 1u;                                    // count=1, user descriptor
  g0.y = lds_off;                               // lds_addr (bytes)
  g0.z = ga_lo;                                 // global_addr[31:0]
  g0.w = (ga_hi & 0x01FFFFFFu) | (2u << 30);    // global_addr[56:32] | type=2
  v8i_ g1;
  g1[0] = (int)0x00010000u;                                     // data_size=1 -> 2 bytes
  g1[1] = (int)((tile_w & 0xFFFFu) << 16);                      // tensor_dim0[15:0]
  g1[2] = (int)(((tile_w >> 16) & 0xFFFFu) |
                ((tile_h & 0xFFFFu) << 16));                    // td0 hi | tensor_dim1 lo
  g1[3] = (int)(((tile_h >> 16) & 0xFFFFu) | (tile_w << 16));   // td1 hi | tile_dim0
  g1[4] = (int)(tile_h & 0xFFFFu);                              // tile_dim1 | tile_dim2=0
  g1[5] = (int)(unsigned)(stride_elems & 0xFFFFFFFFull);        // dim0_stride[31:0]
  g1[6] = (int)(unsigned)((stride_elems >> 32) & 0xFFFFull);    // dim0_stride[47:32]
  g1[7] = 0;
  v4i_ z4 = {0, 0, 0, 0};
#if __clang_major__ >= 23
  v8i_ z8 = {0, 0, 0, 0, 0, 0, 0, 0};
  __builtin_amdgcn_tensor_load_to_lds(g0, g1, z4, z4, z8, 0);
#else
  __builtin_amdgcn_tensor_load_to_lds(g0, g1, z4, z4, 0);
#endif
}

__global__ void __launch_bounds__(256) cvt_f32_bf16(const float* __restrict__ s,
                                                    __hip_bfloat16* __restrict__ d_, int n) {
  __bf16* d = reinterpret_cast<__bf16*>(d_);
  for (int i = blockIdx.x * blockDim.x + threadIdx.x; i < n; i += gridDim.x * blockDim.x)
    d[i] = (__bf16)s[i];
}

// ---------------------------------------------------------------------------
// QKV projection: qkv[bt, f] = sum_c x[bt, c] * w_qkv[f, c]
// M = 8192, N = 3072, K = 1024. Block = 8 waves (2x4), wave tile 64x64.
// Epilogue scatters to q[B,H,T,Dh] (scaled 1/sqrt(Dh)), k[B,H,T,Dh], vT[B,H,Dh,T].
// ---------------------------------------------------------------------------
__global__ void __launch_bounds__(256) gemm_qkv(const __hip_bfloat16* __restrict__ xb_,
                                                const __hip_bfloat16* __restrict__ wb_,
                                                __hip_bfloat16* __restrict__ qb_,
                                                __hip_bfloat16* __restrict__ kb_,
                                                __hip_bfloat16* __restrict__ vtb_) {
  const __bf16* xb = reinterpret_cast<const __bf16*>(xb_);
  const __bf16* wb = reinterpret_cast<const __bf16*>(wb_);
  __bf16* qb  = reinterpret_cast<__bf16*>(qb_);
  __bf16* kb  = reinterpret_cast<__bf16*>(kb_);
  __bf16* vtb = reinterpret_cast<__bf16*>(vtb_);

  const int lane = threadIdx.x & 31;
  const int w    = threadIdx.x >> 5;                 // 8 waves: 2 (M) x 4 (N)
  const int row0 = blockIdx.y * 128 + (w >> 2) * 64;
  const int col0 = blockIdx.x * 256 + (w & 3) * 64;

  v8f acc[4][4];
#pragma unroll
  for (int mi = 0; mi < 4; ++mi) {
#pragma unroll
    for (int ni = 0; ni < 4; ++ni) acc[mi][ni] = zero8();
  }

  for (int k0 = 0; k0 < Cc; k0 += 32) {
    if (k0 + 32 < Cc) {
      __builtin_prefetch(xb + (size_t)row0 * Cc + k0 + 32, 0, 1);
      __builtin_prefetch(wb + (size_t)col0 * Cc + k0 + 32, 0, 1);
    }
    v16bf a[4], b[4];
#pragma unroll
    for (int mi = 0; mi < 4; ++mi)
      a[mi] = load_a16x32(xb + (size_t)(row0 + mi * 16) * Cc + k0, Cc, lane);
#pragma unroll
    for (int ni = 0; ni < 4; ++ni)
      b[ni] = load_b32x16(wb + (size_t)(col0 + ni * 16) * Cc + k0, Cc, lane);
#pragma unroll
    for (int mi = 0; mi < 4; ++mi) {
#pragma unroll
      for (int ni = 0; ni < 4; ++ni) acc[mi][ni] = wmma_bf16(a[mi], b[ni], acc[mi][ni]);
    }
  }

  const int n  = lane & 15;
  const int hi = (lane >> 4) & 1;
#pragma unroll
  for (int mi = 0; mi < 4; ++mi) {
#pragma unroll
    for (int ni = 0; ni < 4; ++ni) {
      const int f = col0 + ni * 16 + n;
#pragma unroll
      for (int r = 0; r < 8; ++r) {
        const int grow = row0 + mi * 16 + r + 8 * hi;   // bt index
        const int bidx = grow >> 11;                    // / T
        const int t    = grow & (Tt - 1);
        const float v  = acc[mi][ni][r];
        if (f < Cc) {                                   // Q (pre-scaled)
          const int h = f >> 6, d = f & 63;
          qb[(((size_t)(bidx * Hh + h)) * Tt + t) * DHd + d] = (__bf16)(v * 0.125f);
        } else if (f < 2 * Cc) {                        // K
          const int f2 = f - Cc, h = f2 >> 6, d = f2 & 63;
          kb[(((size_t)(bidx * Hh + h)) * Tt + t) * DHd + d] = (__bf16)v;
        } else {                                        // V, stored transposed [Dh][T]
          const int f2 = f - 2 * Cc, h = f2 >> 6, d = f2 & 63;
          vtb[(((size_t)(bidx * Hh + h)) * DHd + d) * Tt + t] = (__bf16)v;
        }
      }
    }
  }
}

// ---------------------------------------------------------------------------
// Flash attention: one wave handles 16 queries of one (b,h); key blocks of 64.
// K/V tiles are double-buffered through LDS by the Tensor Data Mover: block
// i+1's tiles are DMA'd while block i computes; s_wait_tensorcnt(2) (per-wave
// TDM completions are in-order) gates the current block. S = Q K^T via WMMA,
// online softmax with 16-lane shuffle reductions, P -> LDS -> A-layout,
// O += P V (WMMA).
// 4 waves/block. Dynamic LDS (136KB): [0,8K) P, [8K,72K) K x2buf, [72K,136K) V x2buf.
// ---------------------------------------------------------------------------
__global__ void __launch_bounds__(128) flash_attn(const __hip_bfloat16* __restrict__ qb_,
                                                  const __hip_bfloat16* __restrict__ kb_,
                                                  const __hip_bfloat16* __restrict__ vtb_,
                                                  const int* __restrict__ am,
                                                  __hip_bfloat16* __restrict__ attb_) {
  const __bf16* qb  = reinterpret_cast<const __bf16*>(qb_);
  const __bf16* kb  = reinterpret_cast<const __bf16*>(kb_);
  const __bf16* vtb = reinterpret_cast<const __bf16*>(vtb_);
  __bf16* attb = reinterpret_cast<__bf16*>(attb_);

  extern __shared__ char smem_raw[];
  __bf16* pbuf  = (__bf16*)smem_raw;                     // 4 waves * 16*64
  __bf16* kbase = (__bf16*)(smem_raw + 8 * 1024);        // 4 waves * 2 bufs * 64*64
  __bf16* vbase = (__bf16*)(smem_raw + 72 * 1024);       // 4 waves * 2 bufs * 64*64

  const int lane = threadIdx.x & 31;
  const int w    = __builtin_amdgcn_readfirstlane(threadIdx.x >> 5);
  const int bh   = blockIdx.y;
  const int bidx = bh / Hh;
  const int h    = bh % Hh;
  const int q0   = blockIdx.x * 64 + w * 16;

  const __bf16* Qh = qb  + (size_t)bh * Tt * DHd;      // [T][Dh]
  const __bf16* Kh = kb  + (size_t)bh * Tt * DHd;      // [T][Dh]
  const __bf16* Vh = vtb + (size_t)bh * DHd * Tt;      // [Dh][T]
  __bf16* pb  = pbuf  + w * 16 * 64;
  __bf16* kt0 = kbase + (w * 2 + 0) * 4096;
  __bf16* kt1 = kbase + (w * 2 + 1) * 4096;
  __bf16* vt0 = vbase + (w * 2 + 0) * 4096;
  __bf16* vt1 = vbase + (w * 2 + 1) * 4096;
  const unsigned ktOff0 = (unsigned)(unsigned long long)kt0;  // LDS byte offset = addr[31:0]
  const unsigned ktOff1 = (unsigned)(unsigned long long)kt1;
  const unsigned vtOff0 = (unsigned)(unsigned long long)vt0;
  const unsigned vtOff1 = (unsigned)(unsigned long long)vt1;

  const int n  = lane & 15;
  const int hi = (lane >> 4) & 1;

  const v16bf aq0 = load_a16x32(Qh + (size_t)q0 * DHd, DHd, lane);
  const v16bf aq1 = load_a16x32(Qh + (size_t)q0 * DHd + 32, DHd, lane);

  v8f o[4] = {zero8(), zero8(), zero8(), zero8()};
  float mrow[8], lrow[8];
#pragma unroll
  for (int r = 0; r < 8; ++r) { mrow[r] = -3.0e38f; lrow[r] = 0.f; }

  const int nblocks = (q0 + 16 + 63) >> 6;             // causal bound
  // ---- prime the TDM pipeline with block 0's K/V tiles ----
  tdm_load_2d(ktOff0, Kh, 64, 64, DHd);
  tdm_load_2d(vtOff0, Vh, 64, 64, Tt);

  for (int ib = 0; ib < nblocks; ++ib) {
    const int j0  = ib << 6;
    const int cur = ib & 1;
    if (ib + 1 < nblocks) {
      // issue next block's DMA into the alternate buffer, then wait for the
      // current block only (in-order TDM: <=2 outstanding => current arrived)
      tdm_load_2d(cur ? ktOff0 : ktOff1, Kh + (size_t)(j0 + 64) * DHd, 64, 64, DHd);
      tdm_load_2d(cur ? vtOff0 : vtOff1, Vh + j0 + 64, 64, 64, Tt);
      __builtin_amdgcn_s_wait_tensorcnt(2);
    } else {
      __builtin_amdgcn_s_wait_tensorcnt(0);
    }
    const __bf16* kt = cur ? kt1 : kt0;
    const __bf16* vt = cur ? vt1 : vt0;

    // ---- scores S = Q K^T for a 16x64 block (B-frags from LDS) ----
    v8f s[4];
#pragma unroll
    for (int jt = 0; jt < 4; ++jt) {
      v16bf bk0 = load_b32x16(kt + jt * 16 * 64, 64, lane);
      v16bf bk1 = load_b32x16(kt + jt * 16 * 64 + 32, 64, lane);
      v8f z = zero8();
      z = wmma_bf16(aq0, bk0, z);
      s[jt] = wmma_bf16(aq1, bk1, z);
    }
    // ---- causal + attn_mask masking (finite -inf avoids NaN rows) ----
#pragma unroll
    for (int jt = 0; jt < 4; ++jt) {
      const int j = j0 + jt * 16 + n;                  // this lane's key column
      const bool kv = (am[bidx * Tt + j] != 0);
#pragma unroll
      for (int r = 0; r < 8; ++r) {
        const int i = q0 + r + 8 * hi;                 // this element's query row
        if (j > i || !kv) s[jt][r] = -3.0e38f;
      }
    }
    // ---- online softmax ----
    float alpha[8];
#pragma unroll
    for (int r = 0; r < 8; ++r) {
      float mx = fmaxf(fmaxf(s[0][r], s[1][r]), fmaxf(s[2][r], s[3][r]));
      mx = rmax16(mx);
      const float mn = fmaxf(mrow[r], mx);
      alpha[r] = __expf(mrow[r] - mn);
      mrow[r] = mn;
    }
    float rs[8];
#pragma unroll
    for (int r = 0; r < 8; ++r) rs[r] = 0.f;
#pragma unroll
    for (int jt = 0; jt < 4; ++jt) {
#pragma unroll
      for (int r = 0; r < 8; ++r) {
        const float p = __expf(s[jt][r] - mrow[r]);
        s[jt][r] = p;
        rs[r] += p;
      }
    }
#pragma unroll
    for (int r = 0; r < 8; ++r) lrow[r] = lrow[r] * alpha[r] + rsum16(rs[r]);
#pragma unroll
    for (int dt = 0; dt < 4; ++dt) {
#pragma unroll
      for (int r = 0; r < 8; ++r) o[dt][r] *= alpha[r];
    }
    // ---- P (C-layout) -> LDS row-major bf16, reload in A-layout ----
#pragma unroll
    for (int jt = 0; jt < 4; ++jt) {
#pragma unroll
      for (int r = 0; r < 8; ++r) pb[(r + 8 * hi) * 64 + jt * 16 + n] = (__bf16)s[jt][r];
    }
    asm volatile("s_wait_dscnt 0" ::: "memory");       // wave-local LDS RAW
    const v16bf pa0 = load_a16x32(pb, 64, lane);
    const v16bf pa1 = load_a16x32(pb + 32, 64, lane);
    // ---- O += P V  (V^T tile in LDS: columns d, K-contiguous along keys) ----
#pragma unroll
    for (int dt = 0; dt < 4; ++dt) {
      v16bf bv0 = load_b32x16(vt + dt * 16 * 64, 64, lane);
      v16bf bv1 = load_b32x16(vt + dt * 16 * 64 + 32, 64, lane);
      o[dt] = wmma_bf16(pa0, bv0, o[dt]);
      o[dt] = wmma_bf16(pa1, bv1, o[dt]);
    }
  }

  // ---- normalize and write attention output as bf16 [B*T, C] ----
#pragma unroll
  for (int r = 0; r < 8; ++r) {
    const float inv = lrow[r] > 0.f ? 1.f / lrow[r] : 0.f;
    const int t = q0 + r + 8 * hi;
    const size_t rowoff = ((size_t)bidx * Tt + t) * Cc + h * DHd + n;
#pragma unroll
    for (int dt = 0; dt < 4; ++dt) attb[rowoff + dt * 16] = (__bf16)(o[dt][r] * inv);
  }
}

// ---------------------------------------------------------------------------
// Output projection: out[bt, f] = sum_c att[bt, c] * w_out[f, c]   (f32 out)
// ---------------------------------------------------------------------------
__global__ void __launch_bounds__(256) gemm_out(const __hip_bfloat16* __restrict__ ab_,
                                                const __hip_bfloat16* __restrict__ wb_,
                                                float* __restrict__ out) {
  const __bf16* ab = reinterpret_cast<const __bf16*>(ab_);
  const __bf16* wb = reinterpret_cast<const __bf16*>(wb_);

  const int lane = threadIdx.x & 31;
  const int w    = threadIdx.x >> 5;
  const int row0 = blockIdx.y * 128 + (w >> 2) * 64;
  const int col0 = blockIdx.x * 256 + (w & 3) * 64;

  v8f acc[4][4];
#pragma unroll
  for (int mi = 0; mi < 4; ++mi) {
#pragma unroll
    for (int ni = 0; ni < 4; ++ni) acc[mi][ni] = zero8();
  }

  for (int k0 = 0; k0 < Cc; k0 += 32) {
    if (k0 + 32 < Cc) {
      __builtin_prefetch(ab + (size_t)row0 * Cc + k0 + 32, 0, 1);
      __builtin_prefetch(wb + (size_t)col0 * Cc + k0 + 32, 0, 1);
    }
    v16bf a[4], b[4];
#pragma unroll
    for (int mi = 0; mi < 4; ++mi)
      a[mi] = load_a16x32(ab + (size_t)(row0 + mi * 16) * Cc + k0, Cc, lane);
#pragma unroll
    for (int ni = 0; ni < 4; ++ni)
      b[ni] = load_b32x16(wb + (size_t)(col0 + ni * 16) * Cc + k0, Cc, lane);
#pragma unroll
    for (int mi = 0; mi < 4; ++mi) {
#pragma unroll
      for (int ni = 0; ni < 4; ++ni) acc[mi][ni] = wmma_bf16(a[mi], b[ni], acc[mi][ni]);
    }
  }

  const int n  = lane & 15;
  const int hi = (lane >> 4) & 1;
#pragma unroll
  for (int mi = 0; mi < 4; ++mi) {
#pragma unroll
    for (int ni = 0; ni < 4; ++ni) {
#pragma unroll
      for (int r = 0; r < 8; ++r) {
        const int grow = row0 + mi * 16 + r + 8 * hi;
        out[(size_t)grow * Cc + col0 + ni * 16 + n] = acc[mi][ni][r];
      }
    }
  }
}

extern "C" void kernel_launch(void* const* d_in, const int* in_sizes, int n_in,
                              void* d_out, int out_size, void* d_ws, size_t ws_size,
                              hipStream_t stream) {
  const float* x    = (const float*)d_in[0];   // [B,T,C]
  const int*   am   = (const int*)d_in[1];     // [B,T]
  const float* wqkv = (const float*)d_in[2];   // [3C,C]
  const float* wout = (const float*)d_in[3];   // [C,C]
  float* out = (float*)d_out;                  // [B,T,C] f32

  char* ws = (char*)d_ws;
  size_t off = 0;
  auto alloc = [&](size_t bytes) -> void* {
    void* p = ws + off;
    off += (bytes + 255) & ~(size_t)255;
    return p;
  };

  __hip_bfloat16* xb    = (__hip_bfloat16*)alloc((size_t)BT * Cc * 2);
  __hip_bfloat16* wqkvb = (__hip_bfloat16*)alloc((size_t)FF * Cc * 2);
  __hip_bfloat16* woutb = (__hip_bfloat16*)alloc((size_t)Cc * Cc * 2);
  __hip_bfloat16* qb    = (__hip_bfloat16*)alloc((size_t)Bb * Hh * Tt * DHd * 2);
  __hip_bfloat16* kbuf  = (__hip_bfloat16*)alloc((size_t)Bb * Hh * Tt * DHd * 2);
  __hip_bfloat16* vtb   = (__hip_bfloat16*)alloc((size_t)Bb * Hh * DHd * Tt * 2);
  __hip_bfloat16* attb  = (__hip_bfloat16*)alloc((size_t)BT * Cc * 2);
  (void)ws_size; (void)in_sizes; (void)n_in; (void)out_size;

  cvt_f32_bf16<<<2048, 256, 0, stream>>>(x, xb, BT * Cc);
  cvt_f32_bf16<<<1024, 256, 0, stream>>>(wqkv, wqkvb, FF * Cc);
  cvt_f32_bf16<<<512, 256, 0, stream>>>(wout, woutb, Cc * Cc);

  gemm_qkv<<<dim3(FF / 256, BT / 128), 256, 0, stream>>>(xb, wqkvb, qb, kbuf, vtb);
  flash_attn<<<dim3(Tt / 64, Bb * Hh), 128, 136 * 1024, stream>>>(qb, kbuf, vtb, am, attb);
  gemm_out<<<dim3(Cc / 256, BT / 128), 256, 0, stream>>>(attb, woutb, out);
}
